// ASPP_40699110096954
// MI455X (gfx1250) — matrix operand order repaired and checked by reading, not compiled, not run on hardware
//
#include <hip/hip_runtime.h>
#include <hip/hip_bf16.h>

#define CDIM 128
#define KOFF 27
#define DNUM 3

typedef __attribute__((ext_vector_type(16))) __bf16 v16bf;
typedef __attribute__((ext_vector_type(8)))  float  v8f;

union Frag16 { v16bf v; uint4 q[2]; };

// ---------------------------------------------------------------- converts
__global__ void to_bf16_kernel(const float* __restrict__ src,
                               __hip_bfloat16* __restrict__ dst, size_t n) {
  size_t i = (size_t)blockIdx.x * blockDim.x + threadIdx.x;
  size_t stride = (size_t)gridDim.x * blockDim.x;
  for (; i < n; i += stride) dst[i] = __float2bfloat16(src[i]);
}

// ---------------------------------------------------------------- sparse conv / GEMM core
// out[i, :] = sum_k fpad[nbr[k,i], :] @ W[k]   (+ addvec broadcast if non-null)
// fpad has Nv+1 rows; row Nv is a zero sentinel for missing neighbors.
// 64-voxel tile per block; double-buffered A (gather) and B (weights) staged with
// CDNA5 async-to-LDS loads, pipelined against the WMMA chunk loop.
// If statsOut != null, also accumulates per-channel sum/sumsq (fused BN stats).
template<bool GATHER>
__global__ __launch_bounds__(256)
void sconv_wmma_kernel(const __hip_bfloat16* __restrict__ fpad,  // [(Nv+1) x 128] bf16
                       const __hip_bfloat16* __restrict__ W,     // [nk x 128 x 128] bf16, ci-major
                       const int* __restrict__ nbr,              // [nk x Nv] (only if GATHER)
                       const float* __restrict__ addvec,         // [128] or null
                       float* __restrict__ out,                  // [Nv x 128] f32
                       float* __restrict__ statsOut,             // [256] sum/sumsq or null
                       int Nv, int nk) {
  __shared__ __align__(16) __bf16 Als[2][64][136];  // 2 x 17408 B, gathered features
  __shared__ __align__(16) __bf16 Bls[2][32][136];  // 2 x  8704 B, weight chunk (32 ci x 128 co)
  __shared__ float sred[2][CDIM];                   // fused BN-stats block reduction

  const int tid  = threadIdx.x;
  const int lane = tid & 31;
  const int wave = tid >> 5;                // 8 waves, wave32
  const int wrow = (wave & 3) * 16;         // wave's 16-voxel row group
  const int wcol = (wave >> 2) * 64;        // wave's 64-channel column half
  const int tileBase = blockIdx.x * 64;

  v8f acc[4];
  #pragma unroll
  for (int t = 0; t < 4; ++t) acc[t] = (v8f){0,0,0,0,0,0,0,0};

  // A staging: thread owns (voxel row, 32-channel segment) -> 4 async b128 loads
  const int ar   = tid >> 2;                // 0..63
  const int aseg = (tid & 3) * 32;          // channel segment
  const int avox = tileBase + ar;
  // B staging: thread owns (ci row, 16-channel segment) -> 2 async b128 loads
  const int br   = tid >> 3;                // 0..31
  const int bseg = (tid & 7) * 16;

  const uint64_t fb64 = (uint64_t)fpad;
  const uint64_t wb64 = (uint64_t)W;

  auto issueA = [&](int k, int buf) {
    int srcRow;
    if (GATHER) srcRow = (avox < Nv) ? nbr[(size_t)k * Nv + avox] : Nv;
    else        srcRow = (avox < Nv) ? avox : Nv;
    const uint32_t goff = (uint32_t)(((unsigned)srcRow * CDIM + (unsigned)aseg) * 2u);
    #pragma unroll
    for (int j = 0; j < 4; ++j) {
      const uint32_t laddr = (uint32_t)(uintptr_t)&Als[buf][ar][aseg + j * 8];
      asm volatile("global_load_async_to_lds_b128 %0, %1, %2"
                   :: "v"(laddr), "v"(goff + (uint32_t)(j * 16)), "s"(fb64)
                   : "memory");
    }
  };
  auto issueB = [&](int k, int ch, int buf) {
    const uint32_t goff =
        (uint32_t)(((((unsigned)k * CDIM) + (unsigned)(ch * 32 + br)) * CDIM + (unsigned)bseg) * 2u);
    #pragma unroll
    for (int j = 0; j < 2; ++j) {
      const uint32_t laddr = (uint32_t)(uintptr_t)&Bls[buf][br][bseg + j * 8];
      asm volatile("global_load_async_to_lds_b128 %0, %1, %2"
                   :: "v"(laddr), "v"(goff + (uint32_t)(j * 16)), "s"(wb64)
                   : "memory");
    }
  };

  const int arow = wrow + (lane & 15);
  const int kb   = (lane >> 4) * 8;         // A-layout: lanes 16-31 hold K+8 block

  // prime the pipeline
  issueA(0, 0);
  issueB(0, 0, 0);

  int abuf = 0, bbuf = 0;
  const int steps = nk * 4;
  #pragma unroll 1
  for (int s = 0; s < steps; ++s) {
    const int ch = s & 3;
    // staged data for step s becomes visible to the whole block
    asm volatile("s_wait_asynccnt 0x0" ::: "memory");
    __syncthreads();

    // prefetch step s+1 into the other buffers (issued after the barrier, so the
    // buffers being overwritten were last read in step s-1 -> no WAR hazard)
    if (s + 1 < steps) {
      const int k1 = (s + 1) >> 2, ch1 = (s + 1) & 3;
      issueB(k1, ch1, bbuf ^ 1);
      if (ch == 3) issueA(k1, abuf ^ 1);
    }

    // ---- compute step s: A frag (16x32 bf16) x 4 B frags (32x16 bf16) ----
    Frag16 a;
    a.q[0] = *(const uint4*)&Als[abuf][arow][ch * 32 + kb];
    a.q[1] = *(const uint4*)&Als[abuf][arow][ch * 32 + kb + 16];
    #pragma unroll
    for (int t = 0; t < 4; ++t) {
      Frag16 b;
      #pragma unroll
      for (int e = 0; e < 8; ++e) {
        b.v[e]     = Bls[bbuf][kb + e][wcol + t * 16 + (lane & 15)];
        b.v[e + 8] = Bls[bbuf][kb + 16 + e][wcol + t * 16 + (lane & 15)];
      }
      acc[t] = __builtin_amdgcn_wmma_f32_16x16x32_bf16(
          false, a.v, false, b.v, (short)0, acc[t], false, false);
    }
    bbuf ^= 1;
    if (ch == 3) abuf ^= 1;
  }

  // ---- store: C/D layout — VGPR r: M = r + 8*(lane>=16), N = lane&15 ----
  const int rbase = tileBase + wrow + ((lane >> 4) * 8);
  #pragma unroll
  for (int t = 0; t < 4; ++t) {
    const int chn = wcol + t * 16 + (lane & 15);
    const float av = addvec ? addvec[chn] : 0.0f;
    #pragma unroll
    for (int r = 0; r < 8; ++r) {
      const int row = rbase + r;
      if (row < Nv) out[(size_t)row * CDIM + chn] = acc[t][r] + av;
    }
  }

  // ---- fused BN statistics: per-channel sum / sumsq over this tile ----
  if (statsOut) {
    if (tid < CDIM) { sred[0][tid] = 0.0f; sred[1][tid] = 0.0f; }
    __syncthreads();
    #pragma unroll
    for (int t = 0; t < 4; ++t) {
      float s = 0.0f, s2 = 0.0f;
      #pragma unroll
      for (int r = 0; r < 8; ++r) {
        const int row = rbase + r;
        const float v = (row < Nv) ? acc[t][r] : 0.0f;
        s += v; s2 += v * v;
      }
      // lanes l and l^16 hold the two row-halves of the same channel
      s  += __shfl_xor(s, 16, 32);
      s2 += __shfl_xor(s2, 16, 32);
      if (lane < 16) {
        const int chn = wcol + t * 16 + lane;
        atomicAdd(&sred[0][chn], s);    // ds_add_f32
        atomicAdd(&sred[1][chn], s2);
      }
    }
    __syncthreads();
    if (tid < CDIM) {
      atomicAdd(&statsOut[tid],        sred[0][tid]);
      atomicAdd(&statsOut[CDIM + tid], sred[1][tid]);
    }
  }
}

// ---------------------------------------------------------------- BN apply (+ReLU / +accumulate)
__global__ __launch_bounds__(256)
void bn_apply_kernel(const float* __restrict__ h, const float* __restrict__ stats,
                     const float* __restrict__ g, const float* __restrict__ b,
                     __hip_bfloat16* __restrict__ outb, float* __restrict__ acc,
                     int Nv, int do_relu) {
  const float invN = 1.0f / (float)Nv;
  const size_t total = (size_t)Nv * CDIM;
  size_t i = (size_t)blockIdx.x * blockDim.x + threadIdx.x;
  const size_t stride = (size_t)gridDim.x * blockDim.x;
  for (; i < total; i += stride) {
    const int c = (int)(i & (CDIM - 1));
    const float m  = stats[c] * invN;
    const float vr = stats[128 + c] * invN - m * m;
    const float sc = g[c] * rsqrtf(vr + 1e-5f);
    float v = (h[i] - m) * sc + b[c];
    if (do_relu) v = fmaxf(v, 0.0f);
    if (outb) outb[i] = __float2bfloat16(v);
    if (acc)  acc[i] += v;
  }
}

// ---------------------------------------------------------------- x_out = relu(y + x) -> bf16, channel sums
__global__ __launch_bounds__(256)
void finalize1_kernel(const float* __restrict__ yacc, const float* __restrict__ x,
                      __hip_bfloat16* __restrict__ xoutb, float* __restrict__ gsum, int Nv) {
  const int t = threadIdx.x;
  const int c = t & 127;
  const int half = t >> 7;
  float s = 0.0f;
  for (int row = blockIdx.x * 2 + half; row < Nv; row += gridDim.x * 2) {
    const size_t i = (size_t)row * CDIM + c;
    float v = yacc[i] + x[i];
    v = fmaxf(v, 0.0f);
    xoutb[i] = __float2bfloat16(v);
    s += v;
  }
  __shared__ float ls[256];
  ls[t] = s;
  __syncthreads();
  if (t < 128) atomicAdd(&gsum[c], ls[t] + ls[t + 128]);
}

// ---------------------------------------------------------------- gvec[c] = blin[c] + mean(x_out) @ wlin[C:2C]
__global__ void gvec_kernel(const float* __restrict__ gsum, const float* __restrict__ wlin,
                            const float* __restrict__ blin, float* __restrict__ gvec,
                            float invN) {
  const int c = threadIdx.x;
  float s = blin[c];
  for (int j = 0; j < CDIM; ++j)
    s += gsum[j] * invN * wlin[(size_t)(CDIM + j) * CDIM + c];
  gvec[c] = s;
}

// ---------------------------------------------------------------- host orchestration
extern "C" void kernel_launch(void* const* d_in, const int* in_sizes, int n_in,
                              void* d_out, int out_size, void* d_ws, size_t ws_size,
                              hipStream_t stream) {
  const float* x    = (const float*)d_in[0];
  const float* w1   = (const float*)d_in[1];
  const float* w2   = (const float*)d_in[2];
  const float* g1   = (const float*)d_in[3];
  const float* b1   = (const float*)d_in[4];
  const float* g2   = (const float*)d_in[5];
  const float* b2   = (const float*)d_in[6];
  const float* wlin = (const float*)d_in[7];
  const float* blin = (const float*)d_in[8];
  const int*   nbr  = (const int*)d_in[9];

  const int Nv = in_sizes[0] / CDIM;                       // 200000
  const size_t wElems = (size_t)DNUM * KOFF * CDIM * CDIM; // per weight tensor

  char* ws = (char*)d_ws;
  size_t off = 0;
  auto take = [&](size_t bytes) -> char* {
    char* p = ws + off;
    off = (off + bytes + 255) & ~(size_t)255;
    return p;
  };
  __hip_bfloat16* xb  = (__hip_bfloat16*)take((size_t)(Nv + 1) * CDIM * 2); // x bf16 / later x_out bf16
  __hip_bfloat16* hb  = (__hip_bfloat16*)take((size_t)(Nv + 1) * CDIM * 2); // normalized h bf16
  float* yacc         = (float*)take((size_t)Nv * CDIM * 4);
  __hip_bfloat16* wb1 = (__hip_bfloat16*)take(wElems * 2);
  __hip_bfloat16* wb2 = (__hip_bfloat16*)take(wElems * 2);
  __hip_bfloat16* wlb = (__hip_bfloat16*)take((size_t)CDIM * CDIM * 2);
  float* stats        = (float*)take(256 * 4);
  float* gsum         = (float*)take(CDIM * 4);
  float* gvec         = (float*)take(CDIM * 4);
  float* hf = (float*)d_out;   // reuse output buffer as f32 conv scratch

  hipMemsetAsync(yacc, 0, (size_t)Nv * CDIM * 4, stream);
  hipMemsetAsync(xb + (size_t)Nv * CDIM, 0, CDIM * 2, stream);  // zero sentinel rows
  hipMemsetAsync(hb + (size_t)Nv * CDIM, 0, CDIM * 2, stream);

  to_bf16_kernel<<<2048, 256, 0, stream>>>(x, xb, (size_t)Nv * CDIM);
  to_bf16_kernel<<<2048, 256, 0, stream>>>(w1, wb1, wElems);
  to_bf16_kernel<<<2048, 256, 0, stream>>>(w2, wb2, wElems);
  to_bf16_kernel<<<256, 256, 0, stream>>>(wlin, wlb, (size_t)CDIM * CDIM);

  const int convBlocks = (Nv + 63) / 64;
  for (int d = 0; d < DNUM; ++d) {
    const __hip_bfloat16* W1d = wb1 + (size_t)d * KOFF * CDIM * CDIM;
    const __hip_bfloat16* W2d = wb2 + (size_t)d * KOFF * CDIM * CDIM;
    const int* nbrd = nbr + (size_t)d * KOFF * Nv;

    // conv1 with fused BN stats
    hipMemsetAsync(stats, 0, 256 * 4, stream);
    sconv_wmma_kernel<true><<<convBlocks, 256, 0, stream>>>(xb, W1d, nbrd, nullptr,
                                                            hf, stats, Nv, KOFF);
    bn_apply_kernel<<<2048, 256, 0, stream>>>(hf, stats, g1 + d * CDIM, b1 + d * CDIM,
                                              hb, nullptr, Nv, 1);

    // conv2 with fused BN stats
    hipMemsetAsync(stats, 0, 256 * 4, stream);
    sconv_wmma_kernel<true><<<convBlocks, 256, 0, stream>>>(hb, W2d, nbrd, nullptr,
                                                            hf, stats, Nv, KOFF);
    bn_apply_kernel<<<2048, 256, 0, stream>>>(hf, stats, g2 + d * CDIM, b2 + d * CDIM,
                                              nullptr, yacc, Nv, 0);
  }

  hipMemsetAsync(gsum, 0, CDIM * 4, stream);
  finalize1_kernel<<<512, 256, 0, stream>>>(yacc, x, xb, gsum, Nv);
  gvec_kernel<<<1, 128, 0, stream>>>(gsum, wlin, blin, gvec, 1.0f / (float)Nv);
  // final 1x1 conv: out = x_out @ wlin[0:C] + gvec (global branch + bias folded in)
  sconv_wmma_kernel<false><<<convBlocks, 256, 0, stream>>>(xb, wlb, nullptr, gvec,
                                                           (float*)d_out, nullptr, Nv, 1);
}
